// RNNCore_47682726920324
// MI455X (gfx1250) — compile-verified, matrix operand order checked by reading, not run on hardware
//
#include <hip/hip_runtime.h>
#include <hip/hip_bf16.h>
#include <cstdint>

// Problem constants (from the reference)
#define S_LEN   512
#define BATCH   64
#define VOCAB   50000
#define EDIM    400
#define EPAD    416      // pad E to a multiple of 32 for WMMA K-steps
#define HDIM    1152
#define G4H     4608     // 4*H
#define NLAYER  3
#define PAD_IDX 1

typedef __attribute__((ext_vector_type(16))) __bf16 v16bf;
typedef __attribute__((ext_vector_type(8)))  float  v8f;

static constexpr long BSTRIDE = (long)G4H * 32;   // halfs per K-step of packed weights
static constexpr long GOFF    = (long)HDIM * 32;  // halfs between gate blocks

__device__ __forceinline__ __bf16 f2bf(float f) {
  unsigned u = __float_as_uint(f);
  unsigned r = (u + 0x7FFFu + ((u >> 16) & 1u)) >> 16;   // RNE to bf16
  unsigned short s = (unsigned short)r;
  __bf16 out;
  __builtin_memcpy(&out, &s, 2);
  return out;
}

// ---------------- prep kernels ----------------

// Embedding gather + cast to bf16, padded to EPAD columns, PAD row zeroed.
__global__ void embed_cast_kernel(const int* __restrict__ tokens,
                                  const float* __restrict__ emb,
                                  __bf16* __restrict__ x0, long total) {
  long i = (long)blockIdx.x * blockDim.x + threadIdx.x;
  if (i >= total) return;
  int  k   = (int)(i % EPAD);
  long row = i / EPAD;
  int  tok = tokens[row];
  float v = 0.f;
  if (k < EDIM && tok != PAD_IDX) v = emb[(long)tok * EDIM + k];
  x0[i] = f2bf(v);
}

// Repack weight [4H][Kin] (fp32, PyTorch layout) into fragment-ready bf16:
// wt[ks][n][kf] with ks = k/32, kf = k%32  (i.e. [Kpad/32][4608][32]).
// A lane's WMMA B-fragment is then two contiguous 16B chunks per column.
__global__ void pack_wt_kernel(const float* __restrict__ w,
                               __bf16* __restrict__ wt, int Kin, int Kpad) {
  long i = (long)blockIdx.x * blockDim.x + threadIdx.x;
  long total = (long)Kpad * G4H;
  if (i >= total) return;
  int n = (int)(i % G4H);
  int k = (int)(i / G4H);
  float v = (k < Kin) ? w[(long)n * Kin + k] : 0.f;
  wt[((long)(k >> 5) * G4H + n) * 32 + (k & 31)] = f2bf(v);
}

__global__ void bias_kernel(const float* __restrict__ bi,
                            const float* __restrict__ bh,
                            float* __restrict__ bs) {
  int i = blockIdx.x * blockDim.x + threadIdx.x;
  if (i < G4H) bs[i] = bi[i] + bh[i];
}

__global__ void zero_state_kernel(__bf16* __restrict__ h, float* __restrict__ c) {
  int i = blockIdx.x * blockDim.x + threadIdx.x;
  if (i < BATCH * HDIM) { h[i] = f2bf(0.f); c[i] = 0.f; }
}

// ---------------- fused LSTM step ----------------
// One launch per (layer, timestep). Grid = H/32 = 36 blocks of 256 threads.
// Block b owns h-units [32*b, 32*b+32) for all 64 batch rows. Each wave:
// m-tile (16 batch rows) x one 16-unit tile x 4 gates -> 4 f32 accumulators.
//
// Data movement: 4-deep LDS pipeline fed by GLOBAL_LOAD_ASYNC_TO_LDS_B128
// (ASYNCcnt-tracked, in-order completion, no VGPR destinations). With DEPTH=4
// the refill of stage ks+3 targets buffer (ks+3)&3 == buffer of stage ks-1,
// whose readers are all past the single top-of-iteration barrier -> one
// barrier per K-step, refill issued before consumption.
// Per stage: A tile 64x32 bf16 (4 KB, 1 async b128/thread),
//            B tile 4x32x32 bf16 (8 KB, 2 async b128/thread).
// Consumption: ds_load_b128 fragment reads + v_wmma_f32_16x16x32_bf16.
//
// LDS layout (single __shared__ object => group segment offset 0):
//   bytes [0, 16384)     : A stages, 4096 B each
//   bytes [16384, 49152) : B stages, 8192 B each
template <int KIN>
__global__ __launch_bounds__(256, 1)
void lstm_step_kernel(const __bf16* __restrict__ xin,
                      const __bf16* __restrict__ wt_ih,   // [KIN/32][4608][32]
                      const __bf16* __restrict__ wt_hh,   // [H/32][4608][32]
                      const float*  __restrict__ bsum,    // [4H] gate-major
                      const __bf16* __restrict__ h_read,  // [B][H] bf16 (t-1)
                      __bf16*       __restrict__ h_write, // [B][H] bf16 (t)
                      float*        __restrict__ c_buf,   // [B][H] fp32
                      float*        __restrict__ out,     // [S][B][H] fp32
                      __bf16*       __restrict__ xnext,   // [S][B][H] bf16 or null
                      int t, int write_next)
{
  constexpr int NK_IN = KIN / 32;
  constexpr int NK_H  = HDIM / 32;
  constexpr int NK    = NK_IN + NK_H;
  constexpr int DEPTH = 4;

  __shared__ __align__(16) __bf16 SMEM[DEPTH * 2048 + DEPTH * 4096]; // 48 KB

  const int tid   = threadIdx.x;
  const int lane  = tid & 31;                      // wave32
  const int wv    = tid >> 5;                      // 0..7
  const int m0    = (wv & 3) * 16;                 // wave batch-row base
  const int ucb   = (wv >> 2) * 16;                // wave unit-tile base (0/16)
  const int n0    = blockIdx.x * 32;               // block h-unit base
  const int l15   = lane & 15;
  const int khalf = (lane < 16) ? 0 : 8;           // 16-bit WMMA K split

  const v8f vzero = {0.f,0.f,0.f,0.f,0.f,0.f,0.f,0.f};
  v8f acc[4];
  #pragma unroll
  for (int g = 0; g < 4; ++g) acc[g] = vzero;

  const char* xrow0 = (const char*)(xin + (long)t * BATCH * KIN);

  // Issue the async copies for K-step `ks` into LDS stage buffer `st`.
  auto issue_stage = [&](int ks, int st) {
    const char* asrc; long arowb; long kl; const char* bsrc;
    if (ks < NK_IN) {
      kl = ks;          asrc = xrow0;                arowb = (long)KIN * 2;
      bsrc = (const char*)wt_ih;
    } else {
      kl = ks - NK_IN;  asrc = (const char*)h_read;  arowb = (long)HDIM * 2;
      bsrc = (const char*)wt_hh;
    }
    { // A tile: [row][32] bf16, one b128 per thread
      unsigned la = (unsigned)(st * 4096 + tid * 16);
      unsigned long long ga = (unsigned long long)(uintptr_t)
          (asrc + (long)(tid >> 2) * arowb + kl * 64 + (tid & 3) * 16);
      asm volatile("global_load_async_to_lds_b128 %0, %1, off"
                   :: "v"(la), "v"(ga) : "memory");
    }
    #pragma unroll
    for (int j = 0; j < 2; ++j) { // B tile: [gate][unit 0..31][32] bf16
      int lb   = (j * 256 + tid) * 16;   // byte offset within 8 KB tile
      int gate = lb >> 11;               // 2 KB per gate chunk
      int win  = lb & 2047;
      unsigned lB = (unsigned)(DEPTH * 4096 + st * 8192 + lb);
      unsigned long long gb = (unsigned long long)(uintptr_t)
          (bsrc + (kl * BSTRIDE + (long)gate * GOFF + (long)n0 * 32) * 2 + win);
      asm volatile("global_load_async_to_lds_b128 %0, %1, off"
                   :: "v"(lB), "v"(gb) : "memory");
    }
    // keep warming L2 a few stages further out
    __builtin_prefetch(bsrc + (kl * BSTRIDE + (long)n0 * 32) * 2 + DEPTH * BSTRIDE * 2, 0, 1);
  };

  issue_stage(0, 0);
  issue_stage(1, 1);
  issue_stage(2, 2);

  int st = 0;
  for (int ks = 0; ks < NK; ++ks) {
    // Async loads complete in order: leaving 2 (resp. 1/0) stages in flight
    // guarantees stage `ks` has fully landed in LDS.
    int rem = NK - 1 - ks;
    if (rem >= 2)      asm volatile("s_wait_asynccnt 6" ::: "memory");
    else if (rem == 1) asm volatile("s_wait_asynccnt 3" ::: "memory");
    else               asm volatile("s_wait_asynccnt 0" ::: "memory");
    __syncthreads();   // tile ks visible; stage ks-1 readers all done

    // Refill buffer (ks+3)&3 (stage ks-1's buffer) before consuming stage ks.
    int sNext = ks + DEPTH - 1;
    if (sNext < NK) issue_stage(sNext, sNext & 3);

    const __bf16* At_s = SMEM + st * 2048;
    const __bf16* Bt_s = SMEM + DEPTH * 2048 + st * 4096;

    // A fragment: two contiguous 16B chunks of this lane's row.
    union { v16bf v; uint4 q[2]; } af;
    {
      const __bf16* ap = At_s + (m0 + l15) * 32 + khalf;
      af.q[0] = *(const uint4*)(ap);
      af.q[1] = *(const uint4*)(ap + 16);
    }
    #pragma unroll
    for (int g = 0; g < 4; ++g) {
      union { v16bf v; uint4 q[2]; } bf_;
      const __bf16* bp = Bt_s + (g * 32 + ucb + l15) * 32 + khalf;
      bf_.q[0] = *(const uint4*)(bp);
      bf_.q[1] = *(const uint4*)(bp + 16);
      acc[g] = __builtin_amdgcn_wmma_f32_16x16x32_bf16(
          false, af.v, false, bf_.v, (short)0, acc[g], false, false);
    }

    st = (st + 1) & (DEPTH - 1);
  }

  // ---- fused elementwise LSTM update ----
  // C/D layout: VGPR r -> M = r (+8 for lanes 16-31), N = lane&15.
  const int nu = n0 + ucb + l15;                   // this lane's h-unit column
  float bi = bsum[0 * HDIM + nu];
  float bf = bsum[1 * HDIM + nu];
  float bg = bsum[2 * HDIM + nu];
  float bo = bsum[3 * HDIM + nu];
  #pragma unroll
  for (int r = 0; r < 8; ++r) {
    int m = m0 + r + ((lane >= 16) ? 8 : 0);       // batch row
    float gi = acc[0][r] + bi;
    float gf = acc[1][r] + bf;
    float gg = acc[2][r] + bg;
    float go = acc[3][r] + bo;
    float si = 1.f / (1.f + __expf(-gi));
    float sf = 1.f / (1.f + __expf(-gf));
    float so = 1.f / (1.f + __expf(-go));
    float tg = tanhf(gg);
    long  ci = (long)m * HDIM + nu;
    float c  = sf * c_buf[ci] + si * tg;
    float h  = so * tanhf(c);
    c_buf[ci]   = c;
    h_write[ci] = f2bf(h);
    out[((long)t * BATCH + m) * HDIM + nu] = h;
    if (write_next) xnext[((long)t * BATCH + m) * HDIM + nu] = f2bf(h);
  }
}

// ---------------- host launch ----------------

extern "C" void kernel_launch(void* const* d_in, const int* in_sizes, int n_in,
                              void* d_out, int out_size, void* d_ws, size_t ws_size,
                              hipStream_t stream) {
  const int*   tokens = (const int*)  d_in[0];
  const float* emb    = (const float*)d_in[1];
  const float* w_ih_0 = (const float*)d_in[2];
  const float* w_hh_0 = (const float*)d_in[3];
  const float* b_ih_0 = (const float*)d_in[4];
  const float* b_hh_0 = (const float*)d_in[5];
  const float* w_ih_r = (const float*)d_in[6];
  const float* w_hh_r = (const float*)d_in[7];
  const float* b_ih_r = (const float*)d_in[8];
  const float* b_hh_r = (const float*)d_in[9];
  float* out = (float*)d_out;

  char* ws = (char*)d_ws;
  auto carve = [&](size_t bytes) -> char* {
    char* p = ws; ws += (bytes + 255) & ~(size_t)255; return p;
  };
  __bf16* wt_ih0 = (__bf16*)carve((size_t)EPAD * G4H * 2);
  __bf16* wt_ih1 = (__bf16*)carve((size_t)HDIM * G4H * 2);
  __bf16* wt_ih2 = (__bf16*)carve((size_t)HDIM * G4H * 2);
  __bf16* wt_hh0 = (__bf16*)carve((size_t)HDIM * G4H * 2);
  __bf16* wt_hh1 = (__bf16*)carve((size_t)HDIM * G4H * 2);
  __bf16* wt_hh2 = (__bf16*)carve((size_t)HDIM * G4H * 2);
  float*  bsum   = (float*) carve((size_t)NLAYER * G4H * 4);
  __bf16* x0     = (__bf16*)carve((size_t)S_LEN * BATCH * EPAD * 2);
  __bf16* x1     = (__bf16*)carve((size_t)S_LEN * BATCH * HDIM * 2);
  __bf16* x2     = (__bf16*)carve((size_t)S_LEN * BATCH * HDIM * 2);
  __bf16* hA     = (__bf16*)carve((size_t)BATCH * HDIM * 2);
  __bf16* hB     = (__bf16*)carve((size_t)BATCH * HDIM * 2);
  float*  cbuf   = (float*) carve((size_t)BATCH * HDIM * 4);
  if ((size_t)(ws - (char*)d_ws) > ws_size) return;  // insufficient scratch

  const long w_lh = (long)G4H * HDIM;  // per-layer weight elements (fp32 src)

  long tw0 = (long)EPAD * G4H;
  long twh = (long)HDIM * G4H;
  pack_wt_kernel<<<(unsigned)((tw0 + 255) / 256), 256, 0, stream>>>(w_ih_0, wt_ih0, EDIM, EPAD);
  pack_wt_kernel<<<(unsigned)((twh + 255) / 256), 256, 0, stream>>>(w_hh_0, wt_hh0, HDIM, HDIM);
  pack_wt_kernel<<<(unsigned)((twh + 255) / 256), 256, 0, stream>>>(w_ih_r,         wt_ih1, HDIM, HDIM);
  pack_wt_kernel<<<(unsigned)((twh + 255) / 256), 256, 0, stream>>>(w_ih_r + w_lh,  wt_ih2, HDIM, HDIM);
  pack_wt_kernel<<<(unsigned)((twh + 255) / 256), 256, 0, stream>>>(w_hh_r,         wt_hh1, HDIM, HDIM);
  pack_wt_kernel<<<(unsigned)((twh + 255) / 256), 256, 0, stream>>>(w_hh_r + w_lh,  wt_hh2, HDIM, HDIM);

  bias_kernel<<<(G4H + 255) / 256, 256, 0, stream>>>(b_ih_0,        b_hh_0,        bsum);
  bias_kernel<<<(G4H + 255) / 256, 256, 0, stream>>>(b_ih_r,        b_hh_r,        bsum + G4H);
  bias_kernel<<<(G4H + 255) / 256, 256, 0, stream>>>(b_ih_r + G4H,  b_hh_r + G4H,  bsum + 2 * G4H);

  long te = (long)S_LEN * BATCH * EPAD;
  embed_cast_kernel<<<(unsigned)((te + 255) / 256), 256, 0, stream>>>(tokens, emb, x0, te);

  const __bf16* xin_l[NLAYER]   = {x0, x1, x2};
  const __bf16* wtih_l[NLAYER]  = {wt_ih0, wt_ih1, wt_ih2};
  const __bf16* wthh_l[NLAYER]  = {wt_hh0, wt_hh1, wt_hh2};
  __bf16*       xnext_l[NLAYER] = {x1, x2, nullptr};

  for (int l = 0; l < NLAYER; ++l) {
    zero_state_kernel<<<(BATCH * HDIM + 255) / 256, 256, 0, stream>>>(hA, cbuf);
    float* out_l = out + (size_t)l * S_LEN * BATCH * HDIM;
    for (int t = 0; t < S_LEN; ++t) {
      __bf16* hr = (t & 1) ? hB : hA;
      __bf16* hw = (t & 1) ? hA : hB;
      int wn = xnext_l[l] != nullptr ? 1 : 0;
      if (l == 0) {
        lstm_step_kernel<EPAD><<<HDIM / 32, 256, 0, stream>>>(
            xin_l[l], wtih_l[l], wthh_l[l], bsum + l * G4H,
            hr, hw, cbuf, out_l, xnext_l[l], t, wn);
      } else {
        lstm_step_kernel<HDIM><<<HDIM / 32, 256, 0, stream>>>(
            xin_l[l], wtih_l[l], wthh_l[l], bsum + l * G4H,
            hr, hw, cbuf, out_l, xnext_l[l], t, wn);
      }
    }
  }
}
